// TensorProds_11269994184892
// MI455X (gfx1250) — matrix operand order repaired and checked by reading, not compiled, not run on hardware
//
#include <hip/hip_runtime.h>

// MI455X / gfx1250, wave32, bf16 WMMA (f32 accumulate).
// 64.4 GFLOP over ~503 MB min HBM traffic -> memory-bound (~22us floor) at
// bf16 WMMA rate. Weights (768 KB) are L2-resident (192 MB L2).
// v2: register-pressure-aware loop structure (ks-outer, unroll 1) to stay
// under 256 VGPRs (v1 spilled into VGPR-MSB addressing), packed bf16 multiply
// for the outer-product B operand, float4 staging loads.

typedef __attribute__((ext_vector_type(16))) __bf16 v16bf;
typedef __attribute__((ext_vector_type(8)))  float  v8f;

#define TOK 16      // tokens per workgroup
#define CS  264     // padded LDS column stride in bf16 elems (528B: 16B-aligned, bank-skewed)
#define NT  32768   // B*N tokens
#define DSZ 65536   // 256*256 weight elems

// ---- prep: convert the three f32 weight matrices to bf16 in workspace ----
__global__ __launch_bounds__(256) void wcvt_kernel(const float* __restrict__ Wl,
                                                   const float* __restrict__ Wr,
                                                   const float* __restrict__ Wo,
                                                   __bf16* __restrict__ out) {
  int idx = blockIdx.x * 256 + threadIdx.x;   // 0 .. 3*65536-1
  float v;
  if (idx < DSZ)            v = Wl[idx];
  else if (idx < 2 * DSZ)   v = Wr[idx - DSZ];
  else                      v = Wo[idx - 2 * DSZ];
  out[idx] = (__bf16)v;
}

// A-operand (16x32 bf16): lane = row M (lane&15); lane>=16 selects other K-halves.
// Lane holds K chunks [kb+h*8 .. +7] and [kb+16+h*8 .. +7]  (h = lane>>4).
__device__ inline v16bf load_a_tile(const __bf16* p) {
  v16bf a;
  *(uint4*)&a       = *(const uint4*)p;        // 8 bf16, low K chunk
  *((uint4*)&a + 1) = *(const uint4*)(p + 16); // 8 bf16, K+16 chunk
  return a;
}
// B-operand (32x16 bf16): lane = column N (lane&15); lane holds K chunks
// [kb+h*8 .. +7] and [kb+16+h*8 .. +7]  (h = lane>>4).
__device__ inline v16bf load_b_tile(const __bf16* p) {
  v16bf b;
  *(uint4*)&b       = *(const uint4*)p;
  *((uint4*)&b + 1) = *(const uint4*)(p + 16);
  return b;
}

#define WMMA_BF16(A, B, C) \
  __builtin_amdgcn_wmma_f32_16x16x32_bf16(false, (A), false, (B), (short)0, (C), false, false)

__global__ __launch_bounds__(256) __attribute__((amdgpu_waves_per_eu(4)))
void tensprod_kernel(
    const float* __restrict__ xl_g, const float* __restrict__ xr_g,
    const __bf16* __restrict__ wbf, float* __restrict__ out) {
  extern __shared__ __bf16 lds[];
  __bf16* xBl = lds;                 // [48 cols][CS]  staged x_l (K = input channel)
  __bf16* xBr = xBl + 48 * CS;       // staged x_r
  __bf16* xlo = xBr + 48 * CS;       // [48 cols][CS]  stage-1 result xl (K = r)
  __bf16* xro = xlo + 48 * CS;       // stage-1 result xr
  const __bf16* Wl = wbf;
  const __bf16* Wr = wbf + DSZ;
  const __bf16* Wo = wbf + 2 * DSZ;

  const int  tid = threadIdx.x;
  const long t0  = (long)blockIdx.x * TOK;

  // ---- stage x tiles into LDS, f32 -> bf16, layout [col = 3*tok+i][I] ----
  const float4* xl4 = (const float4*)(xl_g + t0 * 768);  // [tok][I][i] contiguous
  const float4* xr4 = (const float4*)(xr_g + t0 * 768);
  for (int q = tid; q < TOK * 768 / 4; q += 256) {
    float4 a = xl4[q];
    float4 b = xr4[q];
    int e   = q * 4;
    int tl  = e / 768;
    int rem = e - tl * 768;
    int I   = rem / 3;
    int i   = rem - I * 3;
    int cb  = tl * 3;   // column base for this token
    xBl[(cb + i) * CS + I] = (__bf16)a.x;  xBr[(cb + i) * CS + I] = (__bf16)b.x;
    if (++i == 3) { i = 0; ++I; }
    xBl[(cb + i) * CS + I] = (__bf16)a.y;  xBr[(cb + i) * CS + I] = (__bf16)b.y;
    if (++i == 3) { i = 0; ++I; }
    xBl[(cb + i) * CS + I] = (__bf16)a.z;  xBr[(cb + i) * CS + I] = (__bf16)b.z;
    if (++i == 3) { i = 0; ++I; }
    xBl[(cb + i) * CS + I] = (__bf16)a.w;  xBr[(cb + i) * CS + I] = (__bf16)b.w;
  }
  __syncthreads();

  const int w     = tid >> 5;      // wave id: owns output rows 32w..32w+31
  const int lane  = tid & 31;
  const int colin = lane & 15;
  const int khalf = lane >> 4;
  const int rowA  = colin;

  // ---- stage 1: xl = W_l * xB_l, xr = W_r * xB_r  (M=256, N=48, K=256) ----
  {
    const __bf16* WlA0 = Wl + (w * 32      + rowA) * 256 + khalf * 8;
    const __bf16* WlA1 = Wl + (w * 32 + 16 + rowA) * 256 + khalf * 8;
    const __bf16* WrA0 = Wr + (w * 32      + rowA) * 256 + khalf * 8;
    const __bf16* WrA1 = Wr + (w * 32 + 16 + rowA) * 256 + khalf * 8;
    const __bf16* Bl   = xBl + colin * CS + khalf * 8;
    const __bf16* Br   = xBr + colin * CS + khalf * 8;

    v8f accl[2][3] = {};
    v8f accr[2][3] = {};
#pragma unroll 1
    for (int ks = 0; ks < 8; ++ks) {
      const int kb = ks * 32;
      v16bf al0 = load_a_tile(WlA0 + kb);    // each A tile loaded exactly once
      v16bf al1 = load_a_tile(WlA1 + kb);
      v16bf ar0 = load_a_tile(WrA0 + kb);
      v16bf ar1 = load_a_tile(WrA1 + kb);
#pragma unroll
      for (int ct = 0; ct < 3; ++ct) {
        v16bf bl = load_b_tile(Bl + ct * 16 * CS + kb);
        v16bf br = load_b_tile(Br + ct * 16 * CS + kb);
        accl[0][ct] = WMMA_BF16(al0, bl, accl[0][ct]);
        accl[1][ct] = WMMA_BF16(al1, bl, accl[1][ct]);
        accr[0][ct] = WMMA_BF16(ar0, br, accr[0][ct]);
        accr[1][ct] = WMMA_BF16(ar1, br, accr[1][ct]);
      }
    }
    // C layout: vgpr v, lane<16 -> M=v, lane>=16 -> M=v+8; N = lane&15
    const int rb = w * 32 + 8 * khalf;
#pragma unroll
    for (int ct = 0; ct < 3; ++ct) {
      const int col = ct * 16 + colin;
#pragma unroll
      for (int v = 0; v < 8; ++v) {
        xlo[col * CS + rb + v]      = (__bf16)accl[0][ct][v];
        xlo[col * CS + rb + 16 + v] = (__bf16)accl[1][ct][v];
        xro[col * CS + rb + v]      = (__bf16)accr[0][ct][v];
        xro[col * CS + rb + 16 + v] = (__bf16)accr[1][ct][v];
      }
    }
  }
  __syncthreads();

  // ---- stage 3: out = W_o * (xl outer xr)  (M=256, N=144, K=256) ----
  // B operand built on the fly: B[k, col(t,i,j)] = xl[k, t*3+i] * xr[k, t*3+j]
  // 3 groups of 3 column tiles: 48 accumulator VGPRs live per group.
  const __bf16* WoA0 = Wo + (w * 32      + rowA) * 256 + khalf * 8;
  const __bf16* WoA1 = Wo + (w * 32 + 16 + rowA) * 256 + khalf * 8;
#pragma unroll 1
  for (int ctg = 0; ctg < 3; ++ctg) {
    const __bf16* pl[3];
    const __bf16* pr[3];
    int tls[3], rems[3];
#pragma unroll
    for (int c = 0; c < 3; ++c) {
      int col9 = (ctg * 3 + c) * 16 + colin;
      int tl   = col9 / 9;
      int rem  = col9 - tl * 9;
      int ii   = rem / 3;
      int jj   = rem - ii * 3;
      pl[c]    = xlo + (tl * 3 + ii) * CS + khalf * 8;
      pr[c]    = xro + (tl * 3 + jj) * CS + khalf * 8;
      tls[c]   = tl;
      rems[c]  = rem;
    }
    v8f acc[2][3] = {};
#pragma unroll 1
    for (int ks = 0; ks < 8; ++ks) {
      const int kb = ks * 32;
      v16bf a0 = load_a_tile(WoA0 + kb);
      v16bf a1 = load_a_tile(WoA1 + kb);
#pragma unroll
      for (int c = 0; c < 3; ++c) {
        v16bf xa = load_b_tile(pl[c] + kb);
        v16bf xb = load_b_tile(pr[c] + kb);
        v16bf b  = xa * xb;                 // packed bf16 outer-product element
        acc[0][c] = WMMA_BF16(a0, b, acc[0][c]);
        acc[1][c] = WMMA_BF16(a1, b, acc[1][c]);
      }
    }
    // out[t, O, i, j] f32: flat = t*2304 + O*9 + (i*3+j)
    const int Ob = w * 32 + 8 * khalf;
#pragma unroll
    for (int c = 0; c < 3; ++c) {
      float* po = out + (t0 + tls[c]) * 2304 + rems[c];
#pragma unroll
      for (int v = 0; v < 8; ++v) {
        po[(Ob + v) * 9]      = acc[0][c][v];
        po[(Ob + 16 + v) * 9] = acc[1][c][v];
      }
    }
  }
}

extern "C" void kernel_launch(void* const* d_in, const int* in_sizes, int n_in,
                              void* d_out, int out_size, void* d_ws, size_t ws_size,
                              hipStream_t stream) {
  (void)in_sizes; (void)n_in; (void)out_size; (void)ws_size;
  const float* x_l = (const float*)d_in[0];
  const float* x_r = (const float*)d_in[1];
  const float* W_l = (const float*)d_in[2];
  const float* W_r = (const float*)d_in[3];
  const float* W_o = (const float*)d_in[4];
  __bf16* wbf = (__bf16*)d_ws;   // 3 * 65536 bf16 = 384 KB scratch

  wcvt_kernel<<<(3 * DSZ) / 256, 256, 0, stream>>>(W_l, W_r, W_o, wbf);

  const size_t lds_bytes = (size_t)4 * 48 * CS * sizeof(__bf16);  // ~99 KB (WGP has 320 KB)
  hipFuncSetAttribute(reinterpret_cast<const void*>(tensprod_kernel),
                      hipFuncAttributeMaxDynamicSharedMemorySize, (int)lds_bytes);
  tensprod_kernel<<<NT / TOK, 256, lds_bytes, stream>>>(x_l, x_r, wbf, (float*)d_out);
}
